// VectorQuantizer_25074019074083
// MI455X (gfx1250) — matrix operand order repaired and checked by reading, not compiled, not run on hardware
//
#include <hip/hip_runtime.h>

// ---------------------------------------------------------------------------
// VQ-VAE vector quantizer for MI455X (gfx1250, wave32, WMMA + TDM)
//   z:        [8, 512, 64, 64] fp32  -> rows N = 32768, C = 512
//   m:        [8, 64, 64]      fp32  (mask; w = (m == 0))
//   codebook: [8192, 512]      fp32
// Outputs (concatenated): z_q_st [N*C] f32, loss [1] f32, idx [N] f32
// ---------------------------------------------------------------------------

#define NROWS   32768
#define CDIM    512
#define NCODES  8192
#define NWAVES  8
#define KCHUNKS (CDIM / 32)       // 16
#define MBLOCK  128               // rows per block (16 per wave)
#define NSTAGES (NCODES / 32)     // 256 stages of 32 codes
#define BPITCH  520               // LDS row pitch in halfs (1040 B: 4-bank skew)
#define STAGEQW 4096              // 32 rows * 1024 B = 32 KB = 4096 qwords

typedef __attribute__((ext_vector_type(16))) _Float16 v16h;
typedef __attribute__((ext_vector_type(8)))  float    v8f;
typedef unsigned int u32x4 __attribute__((ext_vector_type(4)));
typedef int          i32x4 __attribute__((ext_vector_type(4)));
typedef int          i32x8 __attribute__((ext_vector_type(8)));

union HFrag  { uint4 u[2]; v16h h; };
union HPair  { _Float16 h[2]; unsigned u; };

// ---- Tensor Data Mover: 1-D 32KB load, LDS pad 4 dwords / 256 dwords -----
__device__ __forceinline__ void tdm_stage_load(unsigned lds_off, const _Float16* gsrc) {
    const unsigned long long ga = (unsigned long long)(uintptr_t)gsrc;
    u32x4 g0;
    g0[0] = 1u;                                            // count=1, load, user
    g0[1] = lds_off;                                       // LDS byte address
    g0[2] = (unsigned)ga;                                  // global_addr[31:0]
    g0[3] = (unsigned)((ga >> 32) & 0x01FFFFFFu)           // global_addr[56:32]
          | 0x80000000u;                                   // type=2 (image)
    i32x8 g1;
    g1[0] = (int)((3u << 16)      // data_size = 8 B
          | (1u << 20)            // pad_enable
          | (7u << 22)            // pad_interval: every 256 dwords (1024 B row)
          | (3u << 25));          // pad_amount: 4 dwords (16 B)
    g1[1] = (int)((STAGEQW & 0xFFFF) << 16);   // tensor_dim0[15:0]   @ bits 63:48
    g1[2] = (int)((STAGEQW >> 16) & 0xFFFF);   // tensor_dim0[31:16]  @ bits 79:64
    g1[3] = (int)((STAGEQW & 0xFFFF) << 16);   // tile_dim0           @ bits 127:112
    g1[4] = 0; g1[5] = 0; g1[6] = 0; g1[7] = 0;
    const i32x4 z4 = {0, 0, 0, 0};
#if __clang_major__ >= 23
    const i32x8 z8 = {0, 0, 0, 0, 0, 0, 0, 0};
    __builtin_amdgcn_tensor_load_to_lds(g0, g1, z4, z4, z8, 0);
#else
    __builtin_amdgcn_tensor_load_to_lds(g0, g1, z4, z4, 0);
#endif
}

// --------------------------- phase 0a: init -------------------------------
__global__ void vq_init_kernel(float* acc) {
    acc[0] = 0.0f;   // weighted sum of |z_q - z|^2
    acc[1] = 0.0f;   // sum of weights
}

// ---------------- phase 0b: NCHW fp32 -> [N, C] fp16 transpose ------------
__global__ __launch_bounds__(256) void vq_transpose_z(const float* __restrict__ z,
                                                      _Float16* __restrict__ zh) {
    __shared__ float tile[32][33];
    const int b   = blockIdx.z;
    const int c0  = blockIdx.y * 32;
    const int hw0 = blockIdx.x * 32;
    {
        const int tx = threadIdx.x & 31;      // hw
        const int ty = threadIdx.x >> 5;      // 0..7 (c rows)
        const float* src = z + ((long)b * CDIM + c0) * 4096 + hw0;
#pragma unroll
        for (int i = 0; i < 32; i += 8)
            tile[ty + i][tx] = src[(long)(ty + i) * 4096 + tx];   // tile[c][hw]
    }
    __syncthreads();
    {
        const int j  = threadIdx.x & 15;      // c-pair (c = 2j, 2j+1)
        const int r0 = threadIdx.x >> 4;      // 0..15 (hw row)
        _Float16* dst = zh + ((long)b * 4096 + hw0) * CDIM + c0;
#pragma unroll
        for (int i = 0; i < 32; i += 16) {
            const int hw = r0 + i;
            HPair p;
            p.h[0] = (_Float16)tile[2 * j][hw];
            p.h[1] = (_Float16)tile[2 * j + 1][hw];
            ((unsigned*)(dst + (long)hw * CDIM))[j] = p.u;
        }
    }
}

// ------------- phase 0c: codebook fp32 -> fp16, plus |c|^2 ----------------
__global__ __launch_bounds__(256) void vq_prep_codebook(const float* __restrict__ cb,
                                                        _Float16* __restrict__ cbh,
                                                        float* __restrict__ cnorm) {
    const int row = blockIdx.x;
    const float2* src = (const float2*)(cb + (long)row * CDIM);
    unsigned* dst = (unsigned*)(cbh + (long)row * CDIM);

    const float2 v = src[threadIdx.x];          // 256 threads x 1 pair = 512
    HPair p;
    p.h[0] = (_Float16)v.x;
    p.h[1] = (_Float16)v.y;
    dst[threadIdx.x] = p.u;
    float s = v.x * v.x + v.y * v.y;

#pragma unroll
    for (int off = 16; off; off >>= 1) s += __shfl_xor(s, off, 32);
    __shared__ float sr[NWAVES];
    if ((threadIdx.x & 31) == 0) sr[threadIdx.x >> 5] = s;
    __syncthreads();
    if (threadIdx.x == 0) {
        float t = 0.0f;
#pragma unroll
        for (int w = 0; w < NWAVES; w++) t += sr[w];
        cnorm[row] = t;
    }
}

// --------------- phase 1: WMMA score GEMM + fused argmin ------------------
// Block = 8 waves = 128 rows (16 private rows per wave; A fully in VGPRs).
// Codebook streamed once per block: TDM stages 32 codes (32 KB) into LDS,
// double-buffered, shared by all 8 waves -> 8x less L2 traffic. Padded LDS
// pitch (1040 B) puts consecutive code rows 4 banks apart (conflict-free
// ds_load_b128). Score s = |c|^2 - 2 * <z, c>.
__global__ __launch_bounds__(256) void vq_argmin_kernel(const _Float16* __restrict__ zh,
                                                        const _Float16* __restrict__ cbh,
                                                        const float* __restrict__ cnorm,
                                                        int* __restrict__ idx_out,
                                                        float* __restrict__ idxf_out) {
    __shared__ __align__(16) _Float16 sB[2][32 * BPITCH];   // 2 x 33,280 B

    const int tid  = threadIdx.x;
    const int wave = tid >> 5;
    const int lane = tid & 31;
    const int laneN = lane & 15;
    const int koffA = (lane >> 4) * 16;
    const int koffB = (lane >> 4) * 16;
    const long rowW = (long)blockIdx.x * MBLOCK + wave * 16;   // this wave's rows

    const unsigned sbOff0 = (unsigned)(uintptr_t)&sB[0][0];
    const unsigned sbOff1 = (unsigned)(uintptr_t)&sB[1][0];

    // kick off stage 0 while A-tiles load
    if (wave == 0) tdm_stage_load(sbOff0, cbh);

    // A-tile: this wave's 16 rows x 512 K, fully register-resident
    HFrag aF[KCHUNKS];
    {
        const _Float16* aRow = zh + (rowW + laneN) * CDIM + koffA;
#pragma unroll
        for (int c = 0; c < KCHUNKS; c++) {
            aF[c].u[0] = *(const uint4*)(aRow + c * 32);
            aF[c].u[1] = *(const uint4*)(aRow + c * 32 + 8);
        }
    }

    float minv[8];
    int   mini[8];
#pragma unroll
    for (int v = 0; v < 8; v++) { minv[v] = 3.402823466e38f; mini[v] = 0; }

    for (int t = 0; t < NSTAGES; t++) {
        if (wave == 0) __builtin_amdgcn_s_wait_tensorcnt(0);
        __syncthreads();   // stage t resident; all waves done with buffer t+1's slot
        if (wave == 0 && t + 1 < NSTAGES)
            tdm_stage_load((t & 1) ? sbOff0 : sbOff1,
                           cbh + (size_t)(t + 1) * 32 * CDIM);

        const _Float16* sBb  = &sB[t & 1][0];
        const _Float16* bRow0 = sBb + laneN * BPITCH + koffB;
        const _Float16* bRow1 = bRow0 + 16 * BPITCH;

        v8f acc0 = {}, acc1 = {};
        HFrag bA[2], bB[2];
        bA[0].u[0] = *(const uint4*)(bRow0);
        bA[0].u[1] = *(const uint4*)(bRow0 + 8);
        bB[0].u[0] = *(const uint4*)(bRow1);
        bB[0].u[1] = *(const uint4*)(bRow1 + 8);

#pragma unroll
        for (int c = 1; c < KCHUNKS; c++) {
            const int k0  = c * 32;
            const int nxt = c & 1;
            const int cur = nxt ^ 1;
            bA[nxt].u[0] = *(const uint4*)(bRow0 + k0);
            bA[nxt].u[1] = *(const uint4*)(bRow0 + k0 + 8);
            bB[nxt].u[0] = *(const uint4*)(bRow1 + k0);
            bB[nxt].u[1] = *(const uint4*)(bRow1 + k0 + 8);
            acc0 = __builtin_amdgcn_wmma_f32_16x16x32_f16(
                false, aF[c - 1].h, false, bA[cur].h, (short)0, acc0, false, false);
            acc1 = __builtin_amdgcn_wmma_f32_16x16x32_f16(
                false, aF[c - 1].h, false, bB[cur].h, (short)0, acc1, false, false);
        }
        acc0 = __builtin_amdgcn_wmma_f32_16x16x32_f16(
            false, aF[KCHUNKS - 1].h, false, bA[1].h, (short)0, acc0, false, false);
        acc1 = __builtin_amdgcn_wmma_f32_16x16x32_f16(
            false, aF[KCHUNKS - 1].h, false, bB[1].h, (short)0, acc1, false, false);

        const int   cand0 = t * 32 + laneN;
        const int   cand1 = cand0 + 16;
        const float cn0   = cnorm[cand0];
        const float cn1   = cnorm[cand1];
#pragma unroll
        for (int v = 0; v < 8; v++) {
            float s0 = __builtin_fmaf(-2.0f, acc0[v], cn0);
            float s1 = __builtin_fmaf(-2.0f, acc1[v], cn1);
            if (s0 < minv[v]) { minv[v] = s0; mini[v] = cand0; }
            if (s1 < minv[v]) { minv[v] = s1; mini[v] = cand1; }
        }
    }

    // reduce across the 16 columns (xor over low 4 lane bits keeps the two
    // M-halves of the C/D layout separate); rows are private to this wave.
#pragma unroll
    for (int v = 0; v < 8; v++) {
        float val = minv[v];
        int   ix  = mini[v];
#pragma unroll
        for (int off = 1; off < 16; off <<= 1) {
            float ov = __shfl_xor(val, off, 32);
            int   oi = __shfl_xor(ix,  off, 32);
            if (ov < val || (ov == val && oi < ix)) { val = ov; ix = oi; }
        }
        if (lane == 0)  { idx_out[rowW + v]     = ix; idxf_out[rowW + v]     = (float)ix; }
        if (lane == 16) { idx_out[rowW + v + 8] = ix; idxf_out[rowW + v + 8] = (float)ix; }
    }
}

// ----------------- phase 2: gather z_q + masked loss sum ------------------
__global__ __launch_bounds__(256) void vq_gather_loss(const float* __restrict__ cb,
                                                      const _Float16* __restrict__ zh,
                                                      const float* __restrict__ m,
                                                      const int* __restrict__ idx,
                                                      float* __restrict__ zq_out,
                                                      float* __restrict__ acc) {
    const int row = blockIdx.x;
    const int k   = idx[row];
    const float w = (m[row] == 0.0f) ? 1.0f : 0.0f;
    const float2*   crow = (const float2*)(cb + (long)k * CDIM);
    const unsigned* zrow = (const unsigned*)(zh + (long)row * CDIM);
    float2*         qdst = (float2*)(zq_out + (long)row * CDIM);

    const float2 q = crow[threadIdx.x];       // 256 threads x 1 pair = 512
    HPair zz; zz.u = zrow[threadIdx.x];
    qdst[threadIdx.x] = q;                    // z_q_st == z_q numerically
    const float d0 = q.x - (float)zz.h[0];
    const float d1 = q.y - (float)zz.h[1];
    float local = w * (d0 * d0 + d1 * d1);

#pragma unroll
    for (int off = 16; off; off >>= 1) local += __shfl_xor(local, off, 32);
    __shared__ float sr[NWAVES];
    if ((threadIdx.x & 31) == 0) sr[threadIdx.x >> 5] = local;
    __syncthreads();
    if (threadIdx.x == 0) {
        float s = 0.0f;
#pragma unroll
        for (int wv = 0; wv < NWAVES; wv++) s += sr[wv];
        atomicAdd(&acc[0], s);
        atomicAdd(&acc[1], w);
    }
}

// --------------------------- phase 3: finalize ----------------------------
__global__ void vq_finalize(const float* __restrict__ acc, float* __restrict__ loss) {
    // both loss terms equal in forward value -> (1 + BETA) * sumsq / (sum_w * C)
    const float denom = acc[1] * (float)CDIM;
    *loss = acc[0] * 1.25f / denom;
}

// ---------------------------------------------------------------------------
extern "C" void kernel_launch(void* const* d_in, const int* in_sizes, int n_in,
                              void* d_out, int out_size, void* d_ws, size_t ws_size,
                              hipStream_t stream) {
    const float* z  = (const float*)d_in[0];   // [8,512,64,64]
    const float* m  = (const float*)d_in[1];   // [8,64,64]
    const float* cb = (const float*)d_in[2];   // [8192,512]

    // workspace layout
    char* ws = (char*)d_ws;
    _Float16* zh    = (_Float16*)ws;                                  // 32 MB
    _Float16* cbh   = (_Float16*)(ws + (size_t)NROWS * CDIM * 2);     //  8 MB
    float*    cnorm = (float*)(ws + (size_t)NROWS * CDIM * 2
                                  + (size_t)NCODES * CDIM * 2);       // 32 KB
    int*      idxw  = (int*)((char*)cnorm + (size_t)NCODES * 4);      // 128 KB
    float*    acc   = (float*)((char*)idxw + (size_t)NROWS * 4);      //  8 B

    float* out   = (float*)d_out;
    float* zq    = out;                                // [N*C]
    float* loss  = out + (size_t)NROWS * CDIM;         // [1]
    float* idxf  = loss + 1;                           // [N]

    vq_init_kernel<<<1, 1, 0, stream>>>(acc);
    vq_transpose_z<<<dim3(4096 / 32, CDIM / 32, 8), 256, 0, stream>>>(z, zh);
    vq_prep_codebook<<<NCODES, 256, 0, stream>>>(cb, cbh, cnorm);
    vq_argmin_kernel<<<NROWS / MBLOCK, 256, 0, stream>>>(zh, cbh, cnorm, idxw, idxf);
    vq_gather_loss<<<NROWS, 256, 0, stream>>>(cb, zh, m, idxw, zq, acc);
    vq_finalize<<<1, 1, 0, stream>>>(acc, loss);
}
